// SchNetInteraction_70566312673869
// MI455X (gfx1250) — compile-verified, matrix operand order checked by reading, not run on hardware
//
#include <hip/hip_runtime.h>
#include <hip/hip_bf16.h>

#define NNODES 25000
#define NEDGES 400000
#define HID    256
#define NFILT  256
#define NGAUSS 50
#define KG     64          // gaussian K padded to multiple of 32
#define FCUTOFF 10.0f
#define FPI 3.14159265358979f

typedef _Float16 v16h __attribute__((ext_vector_type(16)));
typedef _Float16 v8h  __attribute__((ext_vector_type(8)));
typedef float    v8f  __attribute__((ext_vector_type(8)));

// ---------------------------------------------------------------------------
// K-index mapping for 16-bit A (16x32) / B (32x16) fragments, ISA 7.12.2:
// lanes 0-15 hold K = {j} j=0..7 (v0-3) and K = {16+j} (v4-7)
// lanes 16-31 hold K = {8+j}            and K = {24+j}
// ---------------------------------------------------------------------------
__device__ __forceinline__ int kmap(int j, int hi) {
  return (j < 8) ? (hi * 8 + j) : (16 + hi * 8 + (j - 8));
}

__device__ __forceinline__ v8f wmma32(v16h a, v16h b, v8f c) {
  return __builtin_amdgcn_wmma_f32_16x16x32_f16(
      /*neg_a=*/false, a, /*neg_b=*/false, b,
      /*c_mod=*/(short)0, c, /*reuse_a=*/false, /*reuse_b=*/false);
}

// A-fragment from an f32 row in global memory (inline f32->f16 convert).
__device__ __forceinline__ v16h a_from_f32row(const float* __restrict__ row,
                                              int kbase, int hi) {
  const float4* p0 = (const float4*)(row + kbase + hi * 8);
  const float4* p1 = (const float4*)(row + kbase + 16 + hi * 8);
  float4 a0 = p0[0], a1 = p0[1], b0 = p1[0], b1 = p1[1];
  v16h a;
  a[0] = (_Float16)a0.x; a[1] = (_Float16)a0.y; a[2] = (_Float16)a0.z; a[3] = (_Float16)a0.w;
  a[4] = (_Float16)a1.x; a[5] = (_Float16)a1.y; a[6] = (_Float16)a1.z; a[7] = (_Float16)a1.w;
  a[8]  = (_Float16)b0.x; a[9]  = (_Float16)b0.y; a[10] = (_Float16)b0.z; a[11] = (_Float16)b0.w;
  a[12] = (_Float16)b1.x; a[13] = (_Float16)b1.y; a[14] = (_Float16)b1.z; a[15] = (_Float16)b1.w;
  return a;
}

// A-fragment from an f16 row in LDS (two aligned 16B ds loads).
__device__ __forceinline__ v16h a_from_lds(const _Float16* row, int kbase, int hi) {
  v8h lo = *(const v8h*)(row + kbase + hi * 8);
  v8h hh = *(const v8h*)(row + kbase + 16 + hi * 8);
  v16h a;
#pragma unroll
  for (int j = 0; j < 8; ++j) { a[j] = lo[j]; a[8 + j] = hh[j]; }
  return a;
}

// B-fragment from pre-converted fragment buffer: 32B contiguous per lane.
__device__ __forceinline__ v16h b_frag(const _Float16* __restrict__ frag,
                                       int kt, int nt, int NT, int lane) {
  return *(const v16h*)(frag + (((size_t)(kt * NT + nt) * 32) + lane) * 16);
}

__device__ __forceinline__ float ssp(float x) {
  // softplus(x) - log(2), numerically stable
  float ax = fabsf(x);
  return fmaxf(x, 0.0f) + __logf(1.0f + __expf(-ax)) - 0.69314718056f;
}

// ---------------------------------------------------------------------------
// Weight matrix [K][Nc] (f32 row-major) -> f16 WMMA B-fragment layout.
// ---------------------------------------------------------------------------
__global__ void __launch_bounds__(32)
wfrag_convert(const float* __restrict__ W, int K, int Nc,
              _Float16* __restrict__ frag) {
  int NT = Nc / 16;
  int tile = blockIdx.x;                 // tile = kt*NT + nt
  int kt = tile / NT, nt = tile % NT;
  int lane = threadIdx.x;
  int hi = lane >> 4, c = lane & 15;
  v16h o;
#pragma unroll
  for (int j = 0; j < 16; ++j) {
    int k = kt * 32 + kmap(j, hi);
    float v = (k < K) ? W[(size_t)k * Nc + (nt * 16 + c)] : 0.0f;
    o[j] = (_Float16)v;
  }
  *(v16h*)(frag + (((size_t)tile * 32) + lane) * 16) = o;
}

// ---------------------------------------------------------------------------
// x = h @ lin1_w  (25000 x 256 x 256, no bias) -> f16 output (gather buffer).
// Block: 8 waves, 32 rows (2 M-tiles) x 256 cols; wave owns 2 column tiles.
// kt-outer loop: A fragments reused across both N-tiles, B across both M-tiles.
// ---------------------------------------------------------------------------
__global__ void __launch_bounds__(256)
node_gemm_lin1(const float* __restrict__ h, const _Float16* __restrict__ fragW,
               _Float16* __restrict__ x) {
  int tid = threadIdx.x;
  int wave = tid >> 5, lane = tid & 31, hi = lane >> 4, lr = lane & 15;
  int rowbase = blockIdx.x * 32;
  int arow0 = min(rowbase + lr,       NNODES - 1);
  int arow1 = min(rowbase + 16 + lr,  NNODES - 1);
  const float* rp0 = h + (size_t)arow0 * HID;
  const float* rp1 = h + (size_t)arow1 * HID;
  const int NT = NFILT / 16;

  v8f acc[2][2] = {};
#pragma unroll
  for (int kt = 0; kt < HID / 32; ++kt) {
    v16h a0 = a_from_f32row(rp0, kt * 32, hi);
    v16h a1 = a_from_f32row(rp1, kt * 32, hi);
#pragma unroll
    for (int t = 0; t < 2; ++t) {
      v16h b = b_frag(fragW, kt, wave * 2 + t, NT, lane);
      acc[0][t] = wmma32(a0, b, acc[0][t]);
      acc[1][t] = wmma32(a1, b, acc[1][t]);
    }
  }
#pragma unroll
  for (int m = 0; m < 2; ++m)
#pragma unroll
    for (int t = 0; t < 2; ++t) {
      int coln = (wave * 2 + t) * 16 + lr;
#pragma unroll
      for (int i = 0; i < 8; ++i) {
        int r = rowbase + m * 16 + i + hi * 8;
        if (r < NNODES) x[(size_t)r * NFILT + coln] = (_Float16)acc[m][t][i];
      }
    }
}

// ---------------------------------------------------------------------------
// Fused edge kernel: 32 edges per block (8 waves, 2 M-tiles).
//  gaussians -> GEMM1(32x64x256)+b1 -> ssp -> LDS ->
//  GEMM2(32x256x256)+b2 -> *cosine-cutoff -> msg = x[row]*W -> atomicAdd agg
// ---------------------------------------------------------------------------
__global__ void __launch_bounds__(256)
edge_kernel(const float* __restrict__ pos, const int* __restrict__ ei,
            const _Float16* __restrict__ x,
            const _Float16* __restrict__ fragW1, const float* __restrict__ b1,
            const _Float16* __restrict__ fragW2, const float* __restrict__ b2,
            float* __restrict__ agg) {
  __shared__ __align__(16) _Float16 sAttr[32][KG];     // 4 KB
  __shared__ __align__(16) _Float16 sHid[32][NFILT];   // 16 KB
  __shared__ float sW[32];
  __shared__ float sC[32];
  __shared__ int sRow[32], sCol[32];

  int tid = threadIdx.x;
  int ebase = blockIdx.x * 32;

  // keep filter-weight fragments hot in L2 (heavily reused by 12.5k blocks)
  __builtin_prefetch(fragW2 + (size_t)(tid & 31) * 16, 0, 3);

  // phase 1: per-edge geometry
  if (tid < 32) {
    int e = ebase + tid;
    int r = ei[e];                 // edge_index[0][e] (source)
    int c = ei[NEDGES + e];        // edge_index[1][e] (target)
    sRow[tid] = r; sCol[tid] = c;
    float dx = pos[r * 3 + 0] - pos[c * 3 + 0];
    float dy = pos[r * 3 + 1] - pos[c * 3 + 1];
    float dz = pos[r * 3 + 2] - pos[c * 3 + 2];
    float w = sqrtf(dx * dx + dy * dy + dz * dz);
    sW[tid] = w;
    sC[tid] = 0.5f * (__cosf(w * (FPI / FCUTOFF)) + 1.0f);
  }
  __syncthreads();

  // phase 2: gaussian smearing into f16 A-tiles (padded K=64); 8 thr/edge
  {
    int e = tid >> 3, sub = tid & 7;
    float w = sW[e];
    const float delta = FCUTOFF / (float)(NGAUSS - 1);
    const float coeff = -0.5f / (delta * delta);
#pragma unroll
    for (int q = 0; q < 8; ++q) {
      int g = sub * 8 + q;
      float v = 0.0f;
      if (g < NGAUSS) { float d = w - (float)g * delta; v = __expf(coeff * d * d); }
      sAttr[e][g] = (_Float16)v;
    }
  }
  __syncthreads();

  int wave = tid >> 5, lane = tid & 31, hi = lane >> 4, lr = lane & 15;
  const int NT = NFILT / 16;

  // GEMM1: hidden = ssp(attr @ mlp_w1 + b1) -> LDS (f16)
  {
    v8f acc[2][2] = {};
#pragma unroll
    for (int kt = 0; kt < KG / 32; ++kt) {
      v16h a0 = a_from_lds(&sAttr[lr][0],      kt * 32, hi);
      v16h a1 = a_from_lds(&sAttr[16 + lr][0], kt * 32, hi);
#pragma unroll
      for (int t = 0; t < 2; ++t) {
        v16h b = b_frag(fragW1, kt, wave * 2 + t, NT, lane);
        acc[0][t] = wmma32(a0, b, acc[0][t]);
        acc[1][t] = wmma32(a1, b, acc[1][t]);
      }
    }
#pragma unroll
    for (int t = 0; t < 2; ++t) {
      float bias = b1[(wave * 2 + t) * 16 + lr];
#pragma unroll
      for (int m = 0; m < 2; ++m)
#pragma unroll
        for (int i = 0; i < 8; ++i)
          sHid[m * 16 + i + hi * 8][(wave * 2 + t) * 16 + lr] =
              (_Float16)ssp(acc[m][t][i] + bias);
    }
  }
  __syncthreads();

  // GEMM2: W = (hidden @ mlp_w2 + b2) * C ; then gather-multiply-scatter
  {
    v8f acc[2][2] = {};
#pragma unroll
    for (int kt = 0; kt < NFILT / 32; ++kt) {
      v16h a0 = a_from_lds(&sHid[lr][0],      kt * 32, hi);
      v16h a1 = a_from_lds(&sHid[16 + lr][0], kt * 32, hi);
#pragma unroll
      for (int t = 0; t < 2; ++t) {
        v16h b = b_frag(fragW2, kt, wave * 2 + t, NT, lane);
        acc[0][t] = wmma32(a0, b, acc[0][t]);
        acc[1][t] = wmma32(a1, b, acc[1][t]);
      }
    }
#pragma unroll
    for (int t = 0; t < 2; ++t) {
      float bias = b2[(wave * 2 + t) * 16 + lr];
      int coln = (wave * 2 + t) * 16 + lr;
#pragma unroll
      for (int m = 0; m < 2; ++m)
#pragma unroll
        for (int i = 0; i < 8; ++i) {
          int e = m * 16 + i + hi * 8;              // edge within block tile
          float Wv = (acc[m][t][i] + bias) * sC[e];
          float xv = (float)x[(size_t)sRow[e] * NFILT + coln];
          atomicAdd(&agg[(size_t)sCol[e] * NFILT + coln], Wv * xv);
        }
    }
  }
}

// ---------------------------------------------------------------------------
// Fused output: h_update = ssp(agg @ lin2_w + lin2_b) @ lin_w + lin_b
// 32 rows per block; intermediate kept as f16 in LDS (never touches HBM).
// ---------------------------------------------------------------------------
__global__ void __launch_bounds__(256)
node_out(const float* __restrict__ agg,
         const _Float16* __restrict__ fragL2, const float* __restrict__ bL2,
         const _Float16* __restrict__ fragL, const float* __restrict__ bL,
         float* __restrict__ out) {
  __shared__ __align__(16) _Float16 sT[32][HID];   // 16 KB
  int tid = threadIdx.x;
  int wave = tid >> 5, lane = tid & 31, hi = lane >> 4, lr = lane & 15;
  int rowbase = blockIdx.x * 32;
  int arow0 = min(rowbase + lr,      NNODES - 1);
  int arow1 = min(rowbase + 16 + lr, NNODES - 1);
  const float* rp0 = agg + (size_t)arow0 * NFILT;
  const float* rp1 = agg + (size_t)arow1 * NFILT;
  const int NT = HID / 16;

  // GEMM A: t = ssp(agg @ lin2_w + b) -> LDS f16
  {
    v8f acc[2][2] = {};
#pragma unroll
    for (int kt = 0; kt < NFILT / 32; ++kt) {
      v16h a0 = a_from_f32row(rp0, kt * 32, hi);
      v16h a1 = a_from_f32row(rp1, kt * 32, hi);
#pragma unroll
      for (int t = 0; t < 2; ++t) {
        v16h b = b_frag(fragL2, kt, wave * 2 + t, NT, lane);
        acc[0][t] = wmma32(a0, b, acc[0][t]);
        acc[1][t] = wmma32(a1, b, acc[1][t]);
      }
    }
#pragma unroll
    for (int t = 0; t < 2; ++t) {
      float bias = bL2[(wave * 2 + t) * 16 + lr];
#pragma unroll
      for (int m = 0; m < 2; ++m)
#pragma unroll
        for (int i = 0; i < 8; ++i)
          sT[m * 16 + i + hi * 8][(wave * 2 + t) * 16 + lr] =
              (_Float16)ssp(acc[m][t][i] + bias);
    }
  }
  __syncthreads();

  // GEMM B: h_update = t @ lin_w + b -> global f32
  {
    v8f acc[2][2] = {};
#pragma unroll
    for (int kt = 0; kt < HID / 32; ++kt) {
      v16h a0 = a_from_lds(&sT[lr][0],      kt * 32, hi);
      v16h a1 = a_from_lds(&sT[16 + lr][0], kt * 32, hi);
#pragma unroll
      for (int t = 0; t < 2; ++t) {
        v16h b = b_frag(fragL, kt, wave * 2 + t, NT, lane);
        acc[0][t] = wmma32(a0, b, acc[0][t]);
        acc[1][t] = wmma32(a1, b, acc[1][t]);
      }
    }
#pragma unroll
    for (int t = 0; t < 2; ++t) {
      float bias = bL[(wave * 2 + t) * 16 + lr];
      int coln = (wave * 2 + t) * 16 + lr;
#pragma unroll
      for (int m = 0; m < 2; ++m)
#pragma unroll
        for (int i = 0; i < 8; ++i) {
          int r = rowbase + m * 16 + i + hi * 8;
          if (r < NNODES) out[(size_t)r * HID + coln] = acc[m][t][i] + bias;
        }
    }
  }
}

__global__ void __launch_bounds__(256)
copy_pos(const float* __restrict__ pos, float* __restrict__ dst, int n) {
  int i = blockIdx.x * blockDim.x + threadIdx.x;
  if (i < n) dst[i] = pos[i];
}

// ---------------------------------------------------------------------------
extern "C" void kernel_launch(void* const* d_in, const int* in_sizes, int n_in,
                              void* d_out, int out_size, void* d_ws, size_t ws_size,
                              hipStream_t stream) {
  const float* h      = (const float*)d_in[0];
  const float* pos    = (const float*)d_in[1];
  const float* lin1_w = (const float*)d_in[2];
  const float* lin2_w = (const float*)d_in[3];
  const float* lin2_b = (const float*)d_in[4];
  const float* mlp_w1 = (const float*)d_in[5];
  const float* mlp_b1 = (const float*)d_in[6];
  const float* mlp_w2 = (const float*)d_in[7];
  const float* mlp_b2 = (const float*)d_in[8];
  const float* lin_w  = (const float*)d_in[9];
  const float* lin_b  = (const float*)d_in[10];
  const int*   eidx   = (const int*)d_in[11];
  float* out = (float*)d_out;

  // workspace carve-up (256B aligned)
  char* ws = (char*)d_ws;
  size_t off = 0;
  auto carve = [&](size_t bytes) -> void* {
    void* p = ws + off;
    off += (bytes + 255) & ~(size_t)255;
    return p;
  };
  _Float16*  x     = (_Float16*)carve(sizeof(_Float16) * (size_t)NNODES * NFILT);
  float*     agg   = (float*)carve(sizeof(float) * (size_t)NNODES * NFILT);
  const size_t fragSz256 = (size_t)(256 / 32) * (256 / 16) * 32 * 16 * sizeof(_Float16);
  const size_t fragSz64  = (size_t)(64 / 32)  * (256 / 16) * 32 * 16 * sizeof(_Float16);
  _Float16* fLin1 = (_Float16*)carve(fragSz256);
  _Float16* fW1   = (_Float16*)carve(fragSz64);
  _Float16* fW2   = (_Float16*)carve(fragSz256);
  _Float16* fL2   = (_Float16*)carve(fragSz256);
  _Float16* fLin  = (_Float16*)carve(fragSz256);

  // 0) convert weights to WMMA fragment layout + zero accumulator
  wfrag_convert<<<(256 / 32) * (256 / 16), 32, 0, stream>>>(lin1_w, 256, 256, fLin1);
  wfrag_convert<<<(64 / 32)  * (256 / 16), 32, 0, stream>>>(mlp_w1, NGAUSS, 256, fW1);
  wfrag_convert<<<(256 / 32) * (256 / 16), 32, 0, stream>>>(mlp_w2, 256, 256, fW2);
  wfrag_convert<<<(256 / 32) * (256 / 16), 32, 0, stream>>>(lin2_w, 256, 256, fL2);
  wfrag_convert<<<(256 / 32) * (256 / 16), 32, 0, stream>>>(lin_w,  256, 256, fLin);
  hipMemsetAsync(agg, 0, sizeof(float) * (size_t)NNODES * NFILT, stream);

  // 1) x = h @ lin1_w (f16 gather buffer)
  int nodeBlocks = (NNODES + 31) / 32;
  node_gemm_lin1<<<nodeBlocks, 256, 0, stream>>>(h, fLin1, x);

  // 2) fused edge MLP + gather-multiply-scatter (dominant kernel)
  edge_kernel<<<NEDGES / 32, 256, 0, stream>>>(pos, eidx, x,
                                               fW1, mlp_b1, fW2, mlp_b2, agg);

  // 3) h_update = ssp(agg @ lin2_w + b) @ lin_w + b
  node_out<<<nodeBlocks, 256, 0, stream>>>(agg, fL2, lin2_b, fLin, lin_b, out);

  // 4) second tuple output: pos passthrough
  int pn = NNODES * 3;
  copy_pos<<<(pn + 255) / 256, 256, 0, stream>>>(pos, out + (size_t)NNODES * HID, pn);
}
